// GQAAttention_41601053229981
// MI455X (gfx1250) — matrix operand order repaired and checked by reading, not compiled, not run on hardware
//
#include <hip/hip_runtime.h>
#include <hip/hip_bf16.h>

typedef __attribute__((ext_vector_type(16))) __bf16 v16bf;
typedef __attribute__((ext_vector_type(8)))  __bf16 v8bf;
typedef __attribute__((ext_vector_type(8)))  float  v8f;
typedef __attribute__((ext_vector_type(4)))  unsigned int v4u;
typedef __attribute__((ext_vector_type(8)))  unsigned int v8u;

#define B_SZ 2
#define SEQ 2048
#define DMODEL 4096
#define NHEADS 32
#define NKV 8
#define HDIM 128
#define QKVDIM 6144   // (32 + 2*8) * 128

static __device__ inline v16bf cat8(v8bf lo, v8bf hi) {
  return __builtin_shufflevector(lo, hi, 0,1,2,3,4,5,6,7,8,9,10,11,12,13,14,15);
}

static __device__ inline v8f wmma_bf16(v16bf a, v16bf b, v8f c) {
  return __builtin_amdgcn_wmma_f32_16x16x32_bf16(
      /*neg_a=*/false, a, /*neg_b=*/false, b,
      /*c_mod=*/(short)0, c, /*reuse_a=*/false, /*reuse_b=*/false);
}

// ---------------------------------------------------------------------------
// TDM: async-DMA one 128x128 bf16 row-major tile (global -> LDS), with LDS
// row padding 128->136 elems done by the TDM pad unit:
//   pad_interval code 5 = 64 DWORDs (= one 128-elem bf16 row)
//   pad_amount  code 3 = 4 DWORDs  (= 8 bf16 pad elems)
// Descriptor per CDNA5 ISA (D# groups 0/1; groups 2/3 NULL
// for a 2D tensor). Issued once per wave; tracked by TENSORcnt (in-order).
// ---------------------------------------------------------------------------
static __device__ inline void tdm_load_tile_128x128_bf16(const __bf16* gsrc,
                                                         unsigned lds_byte_off) {
  const unsigned long long ga = (unsigned long long)(uintptr_t)gsrc;
  v4u g0;
  g0[0] = 1u;                                   // count=1 (valid), user mode
  g0[1] = lds_byte_off;                         // lds_addr
  g0[2] = (unsigned)(ga & 0xFFFFFFFFu);         // global_addr[31:0]
  g0[3] = (unsigned)(ga >> 32) | (2u << 30);    // global_addr[56:32] | type=2
  v8u g1;
  g1[0] = (1u << 16)        // data_size = 2 bytes
        | (1u << 20)        // pad_enable
        | (5u << 22)        // pad_interval = 64 DWORDs
        | (3u << 25);       // pad_amount  = 4 DWORDs
  g1[1] = 128u << 16;       // tensor_dim0 = 128  (bits 79:48)
  g1[2] = 128u << 16;       // tensor_dim1 = 128  (bits 111:80) -- tile == tensor
  g1[3] = 128u << 16;       // tile_dim0 = 128    (bits 127:112)
  g1[4] = 128u;             // tile_dim1 = 128, tile_dim2 = 0
  g1[5] = 128u;             // tensor_dim0_stride = 128 (bits 207:160)
  g1[6] = 0u;
  g1[7] = 0u;
  asm volatile("tensor_load_to_lds %0, %1" :: "s"(g0), "s"(g1) : "memory");
}

// ---------------------------------------------------------------------------
// GEMM: C[M,N] = A[M,K] * B[K,N], fp32 in/out, bf16 WMMA, f32 accum.
// 256 threads (8 waves), 128x128 tile, K-step 32, double-buffered LDS.
// N, K compile-time so all strides are shifts / immediate offsets.
// ---------------------------------------------------------------------------
#define APAD 40   // 32 + 8 bf16 (80B rows, 16B-aligned, conflict-free)
#define BPAD 40

template <int N, int K>
__global__ __launch_bounds__(256)
void gemm_f32_bf16wmma(const float* __restrict__ A, const float* __restrict__ Bm,
                       float* __restrict__ C) {
  __shared__ __align__(16) __bf16 sA[2][128 * APAD];   // [m][k]
  __shared__ __align__(16) __bf16 sB[2][128 * BPAD];   // [n][k] (transposed)

  const int tid  = threadIdx.x;
  const int lane = tid & 31;
  const int wave = tid >> 5;
  const int wm   = wave >> 2;            // 0..1
  const int wn   = wave & 3;             // 0..3
  const int bm   = blockIdx.y * 128;
  const int bn   = blockIdx.x * 128;
  const int r    = lane & 15;
  const int kh   = lane >> 4;

  v8f acc[4][2];
#pragma unroll
  for (int i = 0; i < 4; ++i)
#pragma unroll
    for (int j = 0; j < 2; ++j) acc[i][j] = (v8f)0.0f;

  auto stage = [&](int k0, int buf) {
    // A tile: 128x32 f32 -> bf16, row-major
#pragma unroll
    for (int p = 0; p < 4; ++p) {
      const int row = p * 32 + (tid >> 3);
      const int col = (tid & 7) * 4;
      const float4 v = *(const float4*)(A + (size_t)(bm + row) * K + k0 + col);
      __bf16* d = &sA[buf][row * APAD + col];
      d[0] = (__bf16)v.x; d[1] = (__bf16)v.y; d[2] = (__bf16)v.z; d[3] = (__bf16)v.w;
    }
    // B tile: 32x128 f32 (k x n) -> bf16 transposed [n][k]
#pragma unroll
    for (int p = 0; p < 4; ++p) {
      const int lin = p * 256 + tid;
      const int kk  = lin >> 5;            // 0..31
      const int nn  = (lin & 31) * 4;      // 0..124
      const float4 v = *(const float4*)(Bm + (size_t)(k0 + kk) * N + bn + nn);
      sB[buf][(nn + 0) * BPAD + kk] = (__bf16)v.x;
      sB[buf][(nn + 1) * BPAD + kk] = (__bf16)v.y;
      sB[buf][(nn + 2) * BPAD + kk] = (__bf16)v.z;
      sB[buf][(nn + 3) * BPAD + kk] = (__bf16)v.w;
    }
  };

  stage(0, 0);
  __syncthreads();

  const int nk = K / 32;
  for (int kt = 0; kt < nk; ++kt) {
    const int cur = kt & 1;
    if (kt + 1 < nk) stage((kt + 1) * 32, cur ^ 1);  // overlap with compute

    v16bf af[4], bf[2];
#pragma unroll
    for (int mt = 0; mt < 4; ++mt) {       // A: interleaved K-quarters
      const int row = wm * 64 + mt * 16 + r;
      v8bf lo = *(const v8bf*)&sA[cur][row * APAD + kh * 8];
      v8bf hi = *(const v8bf*)&sA[cur][row * APAD + kh * 8 + 16];
      af[mt] = cat8(lo, hi);
    }
#pragma unroll
    for (int nt = 0; nt < 2; ++nt) {       // B: interleaved K-quarters
      const int col = wn * 32 + nt * 16 + r;
      v8bf lo = *(const v8bf*)&sB[cur][col * BPAD + kh * 8];
      v8bf hi = *(const v8bf*)&sB[cur][col * BPAD + kh * 8 + 16];
      bf[nt] = cat8(lo, hi);
    }
#pragma unroll
    for (int mt = 0; mt < 4; ++mt)
#pragma unroll
      for (int nt = 0; nt < 2; ++nt)
        acc[mt][nt] = wmma_bf16(af[mt], bf[nt], acc[mt][nt]);

    __syncthreads();
  }

  // store: D layout row = kh*8+e, col = r. N is constexpr -> immediate offsets.
  const int row0 = bm + wm * 64 + kh * 8;
  const int col0 = bn + wn * 32 + r;
#pragma unroll
  for (int mt = 0; mt < 4; ++mt)
#pragma unroll
    for (int nt = 0; nt < 2; ++nt) {
      float* p = C + (size_t)(row0 + mt * 16) * N + col0 + nt * 16;
#pragma unroll
      for (int e = 0; e < 8; ++e) p[(size_t)e * N] = acc[mt][nt][e];
    }
}

// ---------------------------------------------------------------------------
// RoPE + split QKV (fp32) -> Q/K/V bf16, head-major. Q pre-scaled by 1/sqrt(D).
// grid.x = B*S*48, block = 128 (one thread per d).
// ---------------------------------------------------------------------------
__global__ __launch_bounds__(128, 4)
void rope_split_kernel(const float* __restrict__ qkv,
                       const float* __restrict__ cosT, const float* __restrict__ sinT,
                       __bf16* __restrict__ Qb, __bf16* __restrict__ Kb,
                       __bf16* __restrict__ Vb) {
  const int idx = blockIdx.x;
  const int h  = idx % 48;
  const int bs = idx / 48;
  const int s  = bs % SEQ;
  const int b  = bs / SEQ;
  const int d  = threadIdx.x;

  const float* base = qkv + (size_t)bs * QKVDIM + h * HDIM;
  float x = base[d];
  if (h < 40) {  // rope applies to q heads (0..31) and k heads (32..39)
    const float other = (d < 64) ? -base[d + 64] : base[d - 64];
    x = x * cosT[s * HDIM + d] + other * sinT[s * HDIM + d];
  }
  if (h < 32) {
    Qb[((size_t)(b * NHEADS + h) * SEQ + s) * HDIM + d] =
        (__bf16)(x * 0.08838834764831845f);       // 1/sqrt(128)
  } else if (h < 40) {
    Kb[((size_t)(b * NKV + (h - 32)) * SEQ + s) * HDIM + d] = (__bf16)x;
  } else {
    Vb[((size_t)(b * NKV + (h - 40)) * SEQ + s) * HDIM + d] = (__bf16)x;
  }
}

// ---------------------------------------------------------------------------
// Flash attention (causal, GQA 4:1). grid = (S/128, B*NHEADS), block = 256.
// K tiles staged by the Tensor Data Mover, double-buffered and pipelined one
// tile ahead (TENSORcnt is in-order: wait<=1 releases the current tile while
// the next DMA stays in flight across the whole compute phase). V transposed
// manually; P spilled to a dedicated per-wave LDS region (no barrier needed:
// same-wave LDS ops are in-order).
// ---------------------------------------------------------------------------
#define KPAD 136   // 128 + 8 (272B rows: 16B-aligned, conflict-free)

__global__ __launch_bounds__(256)
void flash_attn_kernel(const __bf16* __restrict__ Qb, const __bf16* __restrict__ Kb,
                       const __bf16* __restrict__ Vb, float* __restrict__ O) {
  __shared__ __align__(16) __bf16 sK[2][128 * KPAD];  // double-buffered K tiles
  __shared__ __align__(16) __bf16 sVT[128 * KPAD];    // V tile transposed [d][pos]
  __shared__ __align__(16) __bf16 sP[128 * KPAD];     // P (16 rows per wave)

  const int bh   = blockIdx.y;
  const int b    = bh >> 5;
  const int h    = bh & 31;
  const int kvh  = h >> 2;
  const int qbase = blockIdx.x * 128;

  const int tid  = threadIdx.x;
  const int lane = tid & 31;
  const int wave = tid >> 5;
  const int r    = lane & 15;
  const int kh   = lane >> 4;

  // resident Q A-fragments: 16 rows x 128 d per wave
  v16bf qf[4];
  const __bf16* qrow = Qb + ((size_t)(b * NHEADS + h) * SEQ + qbase + wave * 16 + r) * HDIM;
#pragma unroll
  for (int kt = 0; kt < 4; ++kt) {
    v8bf lo = *(const v8bf*)(qrow + kt * 32 + kh * 8);
    v8bf hi = *(const v8bf*)(qrow + kt * 32 + kh * 8 + 16);
    qf[kt] = cat8(lo, hi);
  }

  v8f acc[8];
#pragma unroll
  for (int i = 0; i < 8; ++i) acc[i] = (v8f)0.0f;
  float mI[8], lI[8];
#pragma unroll
  for (int e = 0; e < 8; ++e) { mI[e] = -3.0e38f; lI[e] = 0.0f; }

  const __bf16* Kbase0 = Kb + (size_t)(b * NKV + kvh) * SEQ * HDIM;
  const __bf16* Vbase0 = Vb + (size_t)(b * NKV + kvh) * SEQ * HDIM;

  const int ntile = blockIdx.x + 1;  // causal: kv tiles 0..qtile

  // prime the TDM pipeline with K tile 0
  if (wave == 0)
    tdm_load_tile_128x128_bf16(Kbase0, (unsigned)(uintptr_t)&sK[0][0]);

  for (int jt = 0; jt < ntile; ++jt) {
    const int jbase = jt * 128;
    const int cur   = jt & 1;
    __syncthreads();  // prev-iter readers of sVT / sK[cur^1] are done

    // ---- V tile transposed to [d][pos] (overlaps in-flight TDM) ----
    const __bf16* Vt = Vbase0 + (size_t)jbase * HDIM;
#pragma unroll
    for (int p = 0; p < 8; ++p) {
      const int row = p * 16 + (tid >> 4);
      const int cv  = (tid & 15) * 8;
      v8bf vv = *(const v8bf*)(Vt + (size_t)row * HDIM + cv);
#pragma unroll
      for (int i = 0; i < 8; ++i) sVT[(cv + i) * KPAD + row] = vv[i];
    }
    // prefetch next V tile (one 128B line per thread)
    if (jt + 1 < ntile)
      __builtin_prefetch(Vbase0 + (size_t)(jbase + 128) * HDIM + tid * 64, 0, 1);

    // ---- pipeline: launch K(jt+1) DMA, then release K(jt) ----
    if (wave == 0) {
      if (jt + 1 < ntile) {
        tdm_load_tile_128x128_bf16(Kbase0 + (size_t)(jbase + 128) * HDIM,
                                   (unsigned)(uintptr_t)&sK[cur ^ 1][0]);
        __builtin_amdgcn_s_wait_tensorcnt(1);
      } else {
        __builtin_amdgcn_s_wait_tensorcnt(0);
      }
    }
    __syncthreads();

    // ---- S = Q K^T (scale pre-folded into Q); grouped loads for pipelining ----
    v8f st[8];
#pragma unroll
    for (int i = 0; i < 8; ++i) st[i] = (v8f)0.0f;
#pragma unroll
    for (int kt = 0; kt < 4; ++kt) {
#pragma unroll
      for (int g = 0; g < 2; ++g) {
        v16bf kf[4];
#pragma unroll
        for (int i = 0; i < 4; ++i) {
          const int col = (g * 4 + i) * 16 + r;
          v8bf lo = *(const v8bf*)&sK[cur][col * KPAD + kt * 32 + kh * 8];
          v8bf hi = *(const v8bf*)&sK[cur][col * KPAD + kt * 32 + kh * 8 + 16];
          kf[i] = cat8(lo, hi);
        }
#pragma unroll
        for (int i = 0; i < 4; ++i)
          st[g * 4 + i] = wmma_bf16(qf[kt], kf[i], st[g * 4 + i]);
      }
    }

    // ---- causal mask on diagonal tile ----
    if (jt == blockIdx.x) {
      const int qpos0 = qbase + wave * 16 + kh * 8;
#pragma unroll
      for (int nt = 0; nt < 8; ++nt) {
        const int kpos = jbase + nt * 16 + r;
#pragma unroll
        for (int e = 0; e < 8; ++e)
          if (kpos > qpos0 + e) st[nt][e] = -1.0e30f;
      }
    }

    // ---- online softmax (row = kh*8+e; reduce across the 16-lane half) ----
    float alpha[8];
#pragma unroll
    for (int e = 0; e < 8; ++e) {
      float mx = st[0][e];
#pragma unroll
      for (int nt = 1; nt < 8; ++nt) mx = fmaxf(mx, st[nt][e]);
#pragma unroll
      for (int msk = 1; msk < 16; msk <<= 1) mx = fmaxf(mx, __shfl_xor(mx, msk, 32));
      const float mnew = fmaxf(mI[e], mx);
      alpha[e] = __expf(mI[e] - mnew);
      mI[e] = mnew;
      float rs = 0.0f;
#pragma unroll
      for (int nt = 0; nt < 8; ++nt) {
        st[nt][e] = __expf(st[nt][e] - mnew);
        rs += st[nt][e];
      }
#pragma unroll
      for (int msk = 1; msk < 16; msk <<= 1) rs += __shfl_xor(rs, msk, 32);
      lI[e] = lI[e] * alpha[e] + rs;
    }

    // ---- spill P (16x128 per wave; own region, same-wave in-order LDS) ----
#pragma unroll
    for (int nt = 0; nt < 8; ++nt)
#pragma unroll
      for (int e = 0; e < 8; ++e)
        sP[(wave * 16 + kh * 8 + e) * KPAD + nt * 16 + r] = (__bf16)st[nt][e];

    // ---- rescale accumulator by alpha ----
#pragma unroll
    for (int dt = 0; dt < 8; ++dt)
#pragma unroll
      for (int e = 0; e < 8; ++e) acc[dt][e] *= alpha[e];

    // ---- O += P V (grouped loads for pipelining) ----
#pragma unroll
    for (int kt = 0; kt < 4; ++kt) {
      v8bf plo = *(const v8bf*)&sP[(wave * 16 + r) * KPAD + kt * 32 + kh * 8];
      v8bf phi = *(const v8bf*)&sP[(wave * 16 + r) * KPAD + kt * 32 + kh * 8 + 16];
      const v16bf pf = cat8(plo, phi);
#pragma unroll
      for (int g = 0; g < 2; ++g) {
        v16bf vf[4];
#pragma unroll
        for (int i = 0; i < 4; ++i) {
          const int dcol = (g * 4 + i) * 16 + r;
          v8bf lo = *(const v8bf*)&sVT[dcol * KPAD + kt * 32 + kh * 8];
          v8bf hi = *(const v8bf*)&sVT[dcol * KPAD + kt * 32 + kh * 8 + 16];
          vf[i] = cat8(lo, hi);
        }
#pragma unroll
        for (int i = 0; i < 4; ++i)
          acc[g * 4 + i] = wmma_bf16(pf, vf[i], acc[g * 4 + i]);
      }
    }
  }

  // ---- normalize + store to [b][s][h*128 + d] fp32 ----
#pragma unroll
  for (int dt = 0; dt < 8; ++dt)
#pragma unroll
    for (int e = 0; e < 8; ++e) {
      const int srow = qbase + wave * 16 + kh * 8 + e;
      const int d    = dt * 16 + r;
      O[((size_t)b * SEQ + srow) * DMODEL + h * HDIM + d] = acc[dt][e] / lI[e];
    }
}

// ---------------------------------------------------------------------------
extern "C" void kernel_launch(void* const* d_in, const int* in_sizes, int n_in,
                              void* d_out, int out_size, void* d_ws, size_t ws_size,
                              hipStream_t stream) {
  const float* hidden = (const float*)d_in[0];  // (B,S,4096)
  const float* cosT   = (const float*)d_in[1];  // (S,128)
  const float* sinT   = (const float*)d_in[2];  // (S,128)
  const float* W_qkv  = (const float*)d_in[3];  // (4096,6144)
  const float* W_out  = (const float*)d_in[4];  // (4096,4096)
  float* out = (float*)d_out;

  const int M = B_SZ * SEQ;  // 4096

  // workspace carve-up
  const size_t off_qkv  = 0;
  const size_t sz_qkv   = (size_t)M * QKVDIM * sizeof(float);
  const size_t off_q    = off_qkv + sz_qkv;
  const size_t sz_q     = (size_t)B_SZ * NHEADS * SEQ * HDIM * sizeof(__bf16);
  const size_t off_k    = off_q + sz_q;
  const size_t sz_kv    = (size_t)B_SZ * NKV * SEQ * HDIM * sizeof(__bf16);
  const size_t off_v    = off_k + sz_kv;
  const size_t off_attn = off_v + sz_kv;
  const size_t sz_attn  = (size_t)M * DMODEL * sizeof(float);
  if (ws_size < off_attn + sz_attn) return;

  char* ws = (char*)d_ws;
  float*  qkv  = (float*)(ws + off_qkv);
  __bf16* Qb   = (__bf16*)(ws + off_q);
  __bf16* Kb   = (__bf16*)(ws + off_k);
  __bf16* Vb   = (__bf16*)(ws + off_v);
  float*  attn = (float*)(ws + off_attn);

  // 1. QKV projection: (4096 x 4096) * (4096 x 6144)
  gemm_f32_bf16wmma<QKVDIM, DMODEL><<<dim3(QKVDIM / 128, M / 128), 256, 0, stream>>>(
      hidden, W_qkv, qkv);

  // 2. RoPE + split to bf16 head-major Q/K/V
  rope_split_kernel<<<B_SZ * SEQ * 48, 128, 0, stream>>>(qkv, cosT, sinT, Qb, Kb, Vb);

  // 3. Causal GQA flash attention (pipelined TDM-staged K tiles)
  flash_attn_kernel<<<dim3(SEQ / 128, B_SZ * NHEADS), 256, 0, stream>>>(Qb, Kb, Vb, attn);

  // 4. Output projection: (4096 x 4096) * (4096 x 4096)
  gemm_f32_bf16wmma<DMODEL, DMODEL><<<dim3(DMODEL / 128, M / 128), 256, 0, stream>>>(
      attn, W_out, out);
}
